// MambaTemporal_55911884259811
// MI455X (gfx1250) — matrix operand order repaired and checked by reading, not compiled, not run on hardware
//
#include <hip/hip_runtime.h>
#include <hip/hip_bf16.h>

typedef __attribute__((ext_vector_type(16))) _Float16 v16h;
typedef __attribute__((ext_vector_type(8)))  float    v8f;

#define DEV __device__ __forceinline__

constexpr int kN   = 8;    // batch
constexpr int kT   = 96;   // in_steps == L
constexpr int kC   = 64;   // channels == d_model
constexpr int kV   = 358;  // nodes
constexpr int kL   = 96;
constexpr int kDIN = 128;  // d_inner
constexpr int kOUT = 96;   // out_steps
constexpr float kCLIP = 10000.f;

DEV float safef(float x) {
  if (__builtin_isnan(x)) return 0.f;
  return fminf(fmaxf(x, -kCLIP), kCLIP);
}
DEV float siluf(float x) { return x / (1.f + __expf(-x)); }

// ---- WMMA fragment builders (CDNA5 §7.12.2 layouts, wave32) ----------------
// A fragment: 16x32 (MxK) f16. lane<16 -> K runs [0..7],[16..23];
// lane>=16 -> [8..15],[24..31]; M = lane&15.  (contiguous 16B runs ->
// compiler vectorizes LDS reads into ds_load_b128)
template <typename FA>
DEV v16h frag_a(FA f, int lane) {
  int m  = lane & 15;
  int kh = (lane >> 4) << 3;  // 0 or 8
  v16h a;
#pragma unroll
  for (int j = 0; j < 8; ++j) a[j] = (_Float16)f(m, kh + j);
#pragma unroll
  for (int j = 0; j < 8; ++j) a[8 + j] = (_Float16)f(m, 16 + kh + j);
  return a;
}
// B fragment: 32x16 (KxN) f16. lanes 0-15 hold K=0..7,16..23; lanes 16-31 K=8..15,24..31;
// N = lane&15, half index j<8 -> K = kb + j, j>=8 -> K = 16 + kb + (j-8).
template <typename FB>
DEV v16h frag_b(FB f, int lane) {
  int n  = lane & 15;
  int kb = (lane >> 4) << 3;  // 0 or 8
  v16h b;
#pragma unroll
  for (int j = 0; j < 8; ++j) b[j] = (_Float16)f(kb + j, n);
#pragma unroll
  for (int j = 0; j < 8; ++j) b[8 + j] = (_Float16)f(16 + kb + j, n);
  return b;
}
DEV v8f wmma_f16(v16h a, v16h b, v8f c) {
  return __builtin_amdgcn_wmma_f32_16x16x32_f16(false, a, false, b, (short)0, c,
                                                false, false);
}
// C/D 16x16 f32 tile: element r of v8f -> row = r + 8*(lane>=16), col = lane&15.

__global__ __launch_bounds__(256)
void mamba_temporal_fused(const float* __restrict__ x,       // (8,96,64,358)
                          const float* __restrict__ w_in,    // (256,64)
                          const float* __restrict__ conv_w,  // (128,4)
                          const float* __restrict__ conv_b,  // (128)
                          const float* __restrict__ w_xp,    // (36,128)
                          const float* __restrict__ w_dt,    // (128,4)
                          const float* __restrict__ b_dt,    // (128)
                          const float* __restrict__ A_log,   // (128,16)
                          const float* __restrict__ ssm_D,   // (128)
                          const float* __restrict__ w_out,   // (64,128)
                          const float* __restrict__ w_tp,    // (96,96)
                          const float* __restrict__ b_tp,    // (96)
                          float* __restrict__ out)           // (8,96,64,358)
{
  // LDS staging (~155 KB of 320 KB/WGP -> 2 workgroups per WGP)
  __shared__ _Float16 s_x16[kL * kC];          // 12 KB  f16 input (A of in_proj)
  __shared__ float    s_xin[(kL + 3) * kDIN];  // 49.5 KB  3 zero pad rows + x-branch
  __shared__ _Float16 s_z  [kL * kDIN];        // 24 KB  gate branch (f16)
  __shared__ float    s_xc [kL * kDIN];        // 48 KB  conv+silu; reused as out_proj out
  __shared__ float    s_dbl[kL * 48];          // 18 KB  x_proj: dt(0:4) B(4:20) C(20:36)

  const int tid  = threadIdx.x;
  const int lane = tid & 31;
  const int wave = tid >> 5;              // 8 waves of 32
  const int blk  = blockIdx.x;            // sequence id, 0..2863
  const int n0   = blk / kV;
  const int v0   = blk % kV;

  if (tid == 0) {  // weights are L2-resident; speculative prefetch (global_prefetch_b8)
    __builtin_prefetch(w_in, 0, 1);
    __builtin_prefetch(w_xp, 0, 1);
    __builtin_prefetch(w_out, 0, 1);
    __builtin_prefetch(w_tp, 0, 1);
  }

  // ---- Stage 1: gather x[seq] -> LDS f16; zero the conv pad rows -----------
  for (int i = tid; i < 3 * kDIN; i += 256) s_xin[i] = 0.f;
  for (int i = tid; i < kL * kC; i += 256) {
    int t = i >> 6, c = i & 63;
    s_x16[i] = (_Float16)x[((size_t)(n0 * kT + t) * kC + c) * kV + v0];
  }
  __syncthreads();

  // ---- Stage 2: in_proj  xz(96x256) = x(96x64) @ w_in^T --------------------
  // Split by destination so the store path is loop-invariant (no exec-mask
  // dance around the WMMA results):  tiles 0..47 -> xin (f32), 48..95 -> z (f16)
  for (int tile = wave; tile < 48; tile += 8) {     // tn = 0..7 -> s_xin
    int tm = tile % 6, tn = tile / 6;
    v8f acc = {};
#pragma unroll
    for (int kt = 0; kt < 2; ++kt) {
      v16h a = frag_a([&](int r, int k) -> float {
        return (float)s_x16[(tm * 16 + r) * kC + kt * 32 + k];
      }, lane);
      v16h b = frag_b([&](int k, int nn) -> float {
        return w_in[(tn * 16 + nn) * kC + kt * 32 + k];
      }, lane);
      acc = wmma_f16(a, b, acc);
    }
    int rb = (lane >> 4) << 3, col = tn * 16 + (lane & 15);
#pragma unroll
    for (int r = 0; r < 8; ++r)
      s_xin[(3 + tm * 16 + rb + r) * kDIN + col] = acc[r];
  }
  for (int tile = wave; tile < 48; tile += 8) {     // tn = 8..15 -> s_z
    int tm = tile % 6, tn8 = tile / 6;
    v8f acc = {};
#pragma unroll
    for (int kt = 0; kt < 2; ++kt) {
      v16h a = frag_a([&](int r, int k) -> float {
        return (float)s_x16[(tm * 16 + r) * kC + kt * 32 + k];
      }, lane);
      v16h b = frag_b([&](int k, int nn) -> float {
        return w_in[(128 + tn8 * 16 + nn) * kC + kt * 32 + k];
      }, lane);
      acc = wmma_f16(a, b, acc);
    }
    int rb = (lane >> 4) << 3, col = tn8 * 16 + (lane & 15);
#pragma unroll
    for (int r = 0; r < 8; ++r)
      s_z[(tm * 16 + rb + r) * kDIN + col] = (_Float16)acc[r];
  }
  __syncthreads();

  // ---- Stage 3: causal depthwise conv (k=4) + SiLU (pad rows -> no branch) -
  for (int i = tid; i < kL * kDIN; i += 256) {
    int t = i >> 7, d = i & 127;
    float acc = conv_b[d];
#pragma unroll
    for (int k = 0; k < 4; ++k)
      acc += s_xin[(t + k) * kDIN + d] * conv_w[d * 4 + k];
    s_xc[i] = siluf(acc);
  }
  __syncthreads();

  // ---- Stage 4: x_proj  dbl(96x36) = xc(96x128) @ w_xp^T (N padded to 48) --
  for (int tile = wave; tile < 18; tile += 8) {
    int tm = tile / 3, tn = tile % 3;
    v8f acc = {};
#pragma unroll
    for (int kt = 0; kt < 4; ++kt) {
      v16h a = frag_a([&](int r, int k) -> float {
        return s_xc[(tm * 16 + r) * kDIN + kt * 32 + k];
      }, lane);
      v16h b = frag_b([&](int k, int nn) -> float {
        int e = tn * 16 + nn;
        float w = 0.f;
        if (e < 36) w = w_xp[e * kDIN + kt * 32 + k];
        return w;
      }, lane);
      acc = wmma_f16(a, b, acc);
    }
    int rb = (lane >> 4) << 3, col = tn * 16 + (lane & 15);
#pragma unroll
    for (int r = 0; r < 8; ++r) s_dbl[(tm * 16 + rb + r) * 48 + col] = acc[r];
  }
  __syncthreads();

  // ---- Stage 5: selective-scan, fp32 in registers --------------------------
  // thread -> channel d = tid>>1, states sb..sb+7 (sb = 8*(tid&1))
  float* s_y = s_xin + 3 * kDIN;  // y overwrites xin (dead after conv)
  {
    const int d  = tid >> 1;
    const int sb = (tid & 1) << 3;
    float Arow[8], h[8];
#pragma unroll
    for (int j = 0; j < 8; ++j) {
      Arow[j] = -__expf(A_log[d * 16 + sb + j]);
      h[j] = 0.f;
    }
    float wdt[4];
#pragma unroll
    for (int r = 0; r < 4; ++r) wdt[r] = w_dt[d * 4 + r];
    const float bdt = b_dt[d];
    const float Dd  = ssm_D[d];

    for (int t = 0; t < kL; ++t) {
      const float* dr = &s_dbl[t * 48];
      float s = bdt;
#pragma unroll
      for (int r = 0; r < 4; ++r) s += dr[r] * wdt[r];
      float dtt = (s > 20.f) ? s : log1pf(__expf(s));   // softplus
      float xt  = s_xc[t * kDIN + d];
      float dx  = dtt * xt;
      float acc = 0.f;
#pragma unroll
      for (int j = 0; j < 8; ++j) {
        float Bm = dr[4 + sb + j];
        float Cm = dr[20 + sb + j];
        float dA = __expf(dtt * Arow[j]);
        h[j] = dA * h[j] + dx * Bm;
        acc += h[j] * Cm;
      }
      acc += __shfl_xor(acc, 1, 32);                    // pair-reduce 16 states
      if ((tid & 1) == 0) {
        float z = (float)s_z[t * kDIN + d];
        s_y[t * kDIN + d] = (acc + xt * Dd) * siluf(z);
      }
    }
  }
  __syncthreads();

  // ---- Stage 6: out_proj  yo(96x64) = y(96x128) @ w_out^T, then _safe ------
  float* s_out = s_xc;  // reuse (96x64 f32, stride 64)
  for (int tile = wave; tile < 24; tile += 8) {
    int tm = tile >> 2, tn = tile & 3;
    v8f acc = {};
#pragma unroll
    for (int kt = 0; kt < 4; ++kt) {
      v16h a = frag_a([&](int r, int k) -> float {
        return s_y[(tm * 16 + r) * kDIN + kt * 32 + k];
      }, lane);
      v16h b = frag_b([&](int k, int nn) -> float {
        return w_out[(tn * 16 + nn) * kDIN + kt * 32 + k];
      }, lane);
      acc = wmma_f16(a, b, acc);
    }
    int rb = (lane >> 4) << 3, col = tn * 16 + (lane & 15);
#pragma unroll
    for (int r = 0; r < 8; ++r)
      s_out[(tm * 16 + rb + r) * kC + col] = safef(acc[r]);
  }
  __syncthreads();

  // ---- Stage 7: time_proj  final(96x64) = w_tp(96x96) @ yo(96x64) ----------
  for (int tile = wave; tile < 24; tile += 8) {
    int tm = tile >> 2, tn = tile & 3;
    v8f acc = {};
#pragma unroll
    for (int kt = 0; kt < 3; ++kt) {   // K = 96
      v16h a = frag_a([&](int r, int k) -> float {
        return w_tp[(tm * 16 + r) * kT + kt * 32 + k];
      }, lane);
      v16h b = frag_b([&](int k, int nn) -> float {
        return s_out[(kt * 32 + k) * kC + tn * 16 + nn];
      }, lane);
      acc = wmma_f16(a, b, acc);
    }
    int rb = (lane >> 4) << 3, col = tn * 16 + (lane & 15);
#pragma unroll
    for (int r = 0; r < 8; ++r) {
      int o = tm * 16 + rb + r;
      float val = safef(acc[r] + b_tp[o]);
      out[((size_t)(n0 * kOUT + o) * kC + col) * kV + v0] = val;
    }
  }
}

extern "C" void kernel_launch(void* const* d_in, const int* in_sizes, int n_in,
                              void* d_out, int out_size, void* d_ws, size_t ws_size,
                              hipStream_t stream) {
  (void)in_sizes; (void)n_in; (void)out_size; (void)d_ws; (void)ws_size;
  const float* x      = (const float*)d_in[0];
  const float* w_in   = (const float*)d_in[1];
  const float* conv_w = (const float*)d_in[2];
  const float* conv_b = (const float*)d_in[3];
  const float* w_xp   = (const float*)d_in[4];
  const float* w_dt   = (const float*)d_in[5];
  const float* b_dt   = (const float*)d_in[6];
  const float* A_log  = (const float*)d_in[7];
  const float* ssm_D  = (const float*)d_in[8];
  const float* w_out  = (const float*)d_in[9];
  const float* w_tp   = (const float*)d_in[10];
  const float* b_tp   = (const float*)d_in[11];
  float* o = (float*)d_out;

  dim3 grid(kN * kV);   // 2864 sequences, one workgroup each
  dim3 block(256);      // 8 wave32
  mamba_temporal_fused<<<grid, block, 0, stream>>>(
      x, w_in, conv_w, conv_b, w_xp, w_dt, b_dt, A_log, ssm_D, w_out, w_tp,
      b_tp, o);
}
